// ModelBasicGNNDeeper_54889682043541
// MI455X (gfx1250) — compile-verified, matrix operand order checked by reading, not run on hardware
//
#include <hip/hip_runtime.h>

#define NN 10000      // N_NODES
#define NE 160000     // N_EDGES
#define NG 64         // N_GRAPHS
#define MAXD 1896

typedef __attribute__((ext_vector_type(16))) __bf16          v16bf;
typedef __attribute__((ext_vector_type(16))) unsigned short  v16u16;
typedef __attribute__((ext_vector_type(8)))  unsigned short  v8u16;
typedef __attribute__((ext_vector_type(4)))  unsigned short  v4u16;
typedef __attribute__((ext_vector_type(8)))  float           v8f;

__device__ __forceinline__ float lrelu(float x) { return x > 0.f ? x : 0.1f * x; }

__device__ __forceinline__ unsigned short f2bf(float f) {
    union { float f; unsigned u; } v; v.f = f;
    unsigned u = v.u;
    u += 0x7FFFu + ((u >> 16) & 1u);     // round-to-nearest-even; f2bf(0)=0
    return (unsigned short)(u >> 16);
}

// fragment build (two aligned ds_load_b128 per operand) + 4 WMMA
__device__ __forceinline__ void wmma_step(
    const unsigned short (*As)[40], const unsigned short (*Bt)[40],
    int wm, int wn, int lane, v8f acc[4])
{
    const int m = lane & 15;
    const int h = lane >> 4;
    const unsigned short* arow = &As[wm * 16 + m][0];
    v8u16 alo = *(const v8u16*)(arow + 8 * h);
    v8u16 ahi = *(const v8u16*)(arow + 16 + 8 * h);
    v16bf a = __builtin_bit_cast(v16bf, __builtin_shufflevector(
        alo, ahi, 0, 1, 2, 3, 4, 5, 6, 7, 8, 9, 10, 11, 12, 13, 14, 15));
    #pragma unroll
    for (int t = 0; t < 4; ++t) {
        const unsigned short* brow = &Bt[wn * 64 + t * 16 + m][0];
        v8u16 blo = *(const v8u16*)(brow + 8 * h);
        v8u16 bhi = *(const v8u16*)(brow + 16 + 8 * h);
        v16bf b = __builtin_bit_cast(v16bf, __builtin_shufflevector(
            blo, bhi, 0, 1, 2, 3, 4, 5, 6, 7, 8, 9, 10, 11, 12, 13, 14, 15));
        acc[t] = __builtin_amdgcn_wmma_f32_16x16x32_bf16(
            false, a, false, b, (short)0, acc[t], false, false);
    }
}

// ---------------------------------------------------------------------------
// bf16 WMMA GEMM:  C[M,N] = A[M,K] @ B[K,N]  (f32 in, bf16 tiles, f32 accum)
// fuse==1: C = lrelu(C + bias) (+ residual)
// Block = 256 threads (8 waves). Block tile 64x128, wave tile 16x64 (4 WMMA).
// Main loop = full k-tiles only; partial tail tile hoisted out. Boundary
// handling is branch-free: select the float input to 0 before conversion.
// ---------------------------------------------------------------------------
__global__ __launch_bounds__(256) void gemm_bf16_wmma(
    const float* __restrict__ A, const float* __restrict__ Bw,
    const float* __restrict__ bias, const float* __restrict__ residual,
    float* __restrict__ C, int M, int K, int N, int fuse)
{
    __shared__ unsigned short As[64][40];    // [row][k]   row pitch 80B (16B mult)
    __shared__ unsigned short Bt[128][40];   // [col][k]   transposed B tile

    const int tid  = threadIdx.x;
    const int lane = tid & 31;
    const int wave = tid >> 5;
    const int wm   = wave & 3;      // 4 waves along M (16 rows each)
    const int wn   = wave >> 2;     // 2 waves along N (64 cols each)
    const int rowBase = blockIdx.y * 64;
    const int colBase = blockIdx.x * 128;

    v8f acc[4] = {{}, {}, {}, {}};

    // ---- A staging map: thread -> (row = tid>>2, 8 consecutive k at (tid&3)*8)
    const int ar  = tid >> 2;
    const int ac0 = (tid & 3) << 3;
    const int agr = rowBase + ar;
    const bool aok = agr < M;
    const int asr = min(agr, M - 1);
    const float* ap = A + (size_t)asr * K + ac0;          // advance 32/iter

    // ---- B staging map: 4 quads/thread; quad = 4 consecutive k for one col
    int bc[4], br0[4], bsc[4];
    bool bok[4];
    const float* bp[4];
    #pragma unroll
    for (int i = 0; i < 4; ++i) {
        int q  = tid + i * 256;
        bc[i]  = q & 127;
        br0[i] = (q >> 7) << 2;
        int gc = colBase + bc[i];
        bok[i] = gc < N;
        bsc[i] = min(gc, N - 1);
        bp[i]  = Bw + (size_t)br0[i] * N + bsc[i];        // advance 32*N/iter
    }

    const int kFull = K & ~31;

    // ======== main loop: full k-tiles only, zero boundary logic ========
    for (int k0 = 0; k0 < kFull; k0 += 32) {
        {
            float av[8];
            #pragma unroll
            for (int j = 0; j < 8; ++j) av[j] = ap[j];
            v8u16 pa;
            #pragma unroll
            for (int j = 0; j < 8; ++j)
                pa[j] = f2bf(aok ? av[j] : 0.f);          // branch-free select
            *(v8u16*)&As[ar][ac0] = pa;
        }
        #pragma unroll
        for (int i = 0; i < 4; ++i) {
            float bv[4];
            #pragma unroll
            for (int j = 0; j < 4; ++j) bv[j] = bp[i][(size_t)j * N];
            v4u16 pb;
            #pragma unroll
            for (int j = 0; j < 4; ++j)
                pb[j] = f2bf(bok[i] ? bv[j] : 0.f);       // branch-free select
            *(v4u16*)&Bt[bc[i]][br0[i]] = pb;
        }
        // speculative prefetch of the next tile (drop-on-fault)
        __builtin_prefetch(ap + 32, 0, 3);
        __builtin_prefetch(bp[0] + (size_t)32 * N, 0, 3);
        __builtin_prefetch(bp[2] + (size_t)32 * N, 0, 3);
        __syncthreads();

        wmma_step(As, Bt, wm, wn, lane, acc);
        __syncthreads();

        ap += 32;
        #pragma unroll
        for (int i = 0; i < 4; ++i) bp[i] += (size_t)32 * N;
    }

    // ======== tail: single possibly-partial k-tile ========
    if (kFull < K) {
        const int k0 = kFull;
        {
            float av[8];
            #pragma unroll
            for (int j = 0; j < 8; ++j)
                av[j] = A[(size_t)asr * K + min(k0 + ac0 + j, K - 1)];
            v8u16 pa;
            #pragma unroll
            for (int j = 0; j < 8; ++j) {
                bool ok = aok && (k0 + ac0 + j < K);
                pa[j] = f2bf(ok ? av[j] : 0.f);
            }
            *(v8u16*)&As[ar][ac0] = pa;
        }
        #pragma unroll
        for (int i = 0; i < 4; ++i) {
            float bv[4];
            #pragma unroll
            for (int j = 0; j < 4; ++j)
                bv[j] = Bw[(size_t)min(k0 + br0[i] + j, K - 1) * N + bsc[i]];
            v4u16 pb;
            #pragma unroll
            for (int j = 0; j < 4; ++j) {
                bool ok = bok[i] && (k0 + br0[i] + j < K);
                pb[j] = f2bf(ok ? bv[j] : 0.f);
            }
            *(v4u16*)&Bt[bc[i]][br0[i]] = pb;
        }
        __syncthreads();
        wmma_step(As, Bt, wm, wn, lane, acc);
        // no trailing barrier needed: epilogue reads only registers
    }

    // ---- epilogue: C/D layout lane -> n, VGPR v -> m = v + 8*(lane>=16)
    const int n0 = lane & 15;
    const int hh = lane >> 4;
    #pragma unroll
    for (int t = 0; t < 4; ++t) {
        int gc = colBase + wn * 64 + t * 16 + n0;
        if (gc >= N) continue;
        #pragma unroll
        for (int v = 0; v < 8; ++v) {
            int mrow = rowBase + wm * 16 + v + 8 * hh;
            if (mrow >= M) continue;
            float val = acc[t][v];
            if (fuse) {
                val = lrelu(val + bias[gc]);
                if (residual) val += residual[(size_t)mrow * N + gc];
            }
            C[(size_t)mrow * N + gc] = val;
        }
    }
}

// ---------------------------------------------------------------------------
// CSR build + normalization
// ---------------------------------------------------------------------------
__global__ void init_deg_cnt(float* deg, int* cnt, int n) {
    int i = blockIdx.x * blockDim.x + threadIdx.x;
    if (i < n) { deg[i] = 1.0f; cnt[i] = 0; }   // self-loop weight 1
}

__global__ void count_kernel(const int* __restrict__ col, const float* __restrict__ ew,
                             float* deg, int* cnt, int e) {
    int i = blockIdx.x * blockDim.x + threadIdx.x;
    if (i < e) { atomicAdd(&deg[col[i]], ew[i]); atomicAdd(&cnt[col[i]], 1); }
}

__global__ void dinv_kernel(const float* deg, float* dinv, int n) {
    int i = blockIdx.x * blockDim.x + threadIdx.x;
    if (i < n) { float d = deg[i]; dinv[i] = d > 0.f ? rsqrtf(d) : 0.f; }
}

__global__ void scan_kernel(const int* __restrict__ cnt, int* ptr, int* cursor, int n) {
    __shared__ int part[256];
    int chunk = (n + 255) / 256;
    int s0 = threadIdx.x * chunk;
    int s1 = min(s0 + chunk, n);
    int s = 0;
    for (int i = s0; i < s1; ++i) s += cnt[i];
    part[threadIdx.x] = s;
    __syncthreads();
    if (threadIdx.x == 0) {
        int run = 0;
        for (int i = 0; i < 256; ++i) { int t = part[i]; part[i] = run; run += t; }
        ptr[n] = run;
    }
    __syncthreads();
    int run = part[threadIdx.x];
    for (int i = s0; i < s1; ++i) { ptr[i] = run; cursor[i] = run; run += cnt[i]; }
}

__global__ void fill_kernel(const int* __restrict__ row, const int* __restrict__ col,
                            const float* __restrict__ ew, const float* __restrict__ dinv,
                            int* cursor, int* crow, float* cnrm, int e) {
    int i = blockIdx.x * blockDim.x + threadIdx.x;
    if (i >= e) return;
    int c = col[i], r = row[i];
    int pos = atomicAdd(&cursor[c], 1);
    crow[pos] = r;
    cnrm[pos] = dinv[r] * ew[i] * dinv[c];
}

// ---------------------------------------------------------------------------
// per-node weighted aggregation + bias + leaky-relu (+ residual)
// one block per node; threads stream across feature dim (coalesced).
// float4 path when dout % 4 == 0 (12/492/1896); scalar fallback (86).
// ---------------------------------------------------------------------------
__global__ __launch_bounds__(256) void aggregate_kernel(
    const float* __restrict__ H, const int* __restrict__ ptr,
    const int* __restrict__ rows, const float* __restrict__ nrm,
    const float* __restrict__ dinv, const float* __restrict__ bias,
    const float* __restrict__ residual, float* __restrict__ Y, int dout)
{
    int i = blockIdx.x;
    int b0 = ptr[i], b1 = ptr[i + 1];
    float self = dinv[i] * dinv[i];           // self-loop: dinv*1*dinv

    if ((dout & 3) == 0) {
        int d4 = dout >> 2;
        const float4* __restrict__ H4 = (const float4*)H;
        const float4* __restrict__ B4 = (const float4*)bias;
        const float4* __restrict__ R4 = (const float4*)residual;
        float4* __restrict__ Y4 = (float4*)Y;
        for (int c = threadIdx.x; c < d4; c += blockDim.x) {
            float4 h0 = H4[(size_t)i * d4 + c];
            float ax = self * h0.x, ay = self * h0.y, az = self * h0.z, aw = self * h0.w;
            for (int e = b0; e < b1; ++e) {
                float w = nrm[e];
                float4 hv = H4[(size_t)rows[e] * d4 + c];
                ax += w * hv.x; ay += w * hv.y; az += w * hv.z; aw += w * hv.w;
            }
            float4 bb = B4[c];
            ax = lrelu(ax + bb.x); ay = lrelu(ay + bb.y);
            az = lrelu(az + bb.z); aw = lrelu(aw + bb.w);
            if (residual) {
                float4 rv = R4[(size_t)i * d4 + c];
                ax += rv.x; ay += rv.y; az += rv.z; aw += rv.w;
            }
            float4 o; o.x = ax; o.y = ay; o.z = az; o.w = aw;
            Y4[(size_t)i * d4 + c] = o;
        }
    } else {
        for (int c = threadIdx.x; c < dout; c += blockDim.x) {
            float acc = self * H[(size_t)i * dout + c];
            for (int e = b0; e < b1; ++e)
                acc += nrm[e] * H[(size_t)rows[e] * dout + c];
            acc = lrelu(acc + bias[c]);
            if (residual) acc += residual[(size_t)i * dout + c];
            Y[(size_t)i * dout + c] = acc;
        }
    }
}

__global__ void zero_kernel(float* p, int n) {
    int i = blockIdx.x * blockDim.x + threadIdx.x;
    if (i < n) p[i] = 0.f;
}

__global__ void pool_kernel(const float* __restrict__ h, const int* __restrict__ batch,
                            float* pool, int dout) {
    int i = blockIdx.x;
    int g = batch[i];
    for (int c = threadIdx.x; c < dout; c += blockDim.x)
        atomicAdd(&pool[(size_t)g * dout + c], h[(size_t)i * dout + c]);
}

// ---------------------------------------------------------------------------
extern "C" void kernel_launch(void* const* d_in, const int* in_sizes, int n_in,
                              void* d_out, int out_size, void* d_ws, size_t ws_size,
                              hipStream_t stream)
{
    (void)in_sizes; (void)n_in; (void)out_size; (void)ws_size;

    const float* x0    = (const float*)d_in[0];
    const int*   erow  = (const int*)d_in[1];
    const int*   ecol  = ((const int*)d_in[1]) + NE;
    const float* ew    = (const float*)d_in[2];
    const int*   batch = (const int*)d_in[3];

    const float* convW[8]; const float* convB[8];
    for (int l = 0; l < 8; ++l) { convW[l] = (const float*)d_in[4 + 2 * l]; convB[l] = (const float*)d_in[5 + 2 * l]; }
    const float* fcW[7]; const float* fcB[7];
    for (int l = 0; l < 7; ++l) { fcW[l] = (const float*)d_in[20 + 2 * l]; fcB[l] = (const float*)d_in[21 + 2 * l]; }

    char* ws = (char*)d_ws;
    size_t off = 0;
    auto carve = [&](size_t bytes) -> void* {
        void* p = ws + off; off = (off + bytes + 255) & ~(size_t)255; return p;
    };
    float* deg  = (float*)carve((size_t)NN * 4);
    float* dinv = (float*)carve((size_t)NN * 4);
    int*   cnt  = (int*)  carve((size_t)NN * 4);
    int*   ptr  = (int*)  carve((size_t)(NN + 1) * 4);
    int*   cur  = (int*)  carve((size_t)NN * 4);
    int*   crow = (int*)  carve((size_t)NE * 4);
    float* cnrm = (float*)carve((size_t)NE * 4);
    float* bufA = (float*)carve((size_t)NN * MAXD * 4);
    float* bufB = (float*)carve((size_t)NN * MAXD * 4);
    float* bufH = (float*)carve((size_t)NN * MAXD * 4);
    float* pool = (float*)carve((size_t)NG * MAXD * 4);
    float* fcA  = (float*)carve((size_t)NG * MAXD * 4);
    float* fcBf = (float*)carve((size_t)NG * MAXD * 4);

    // --- CSR + symmetric normalization (once per launch) ---
    init_deg_cnt<<<(NN + 255) / 256, 256, 0, stream>>>(deg, cnt, NN);
    count_kernel<<<(NE + 255) / 256, 256, 0, stream>>>(ecol, ew, deg, cnt, NE);
    dinv_kernel<<<(NN + 255) / 256, 256, 0, stream>>>(deg, dinv, NN);
    scan_kernel<<<1, 256, 0, stream>>>(cnt, ptr, cur, NN);
    fill_kernel<<<(NE + 255) / 256, 256, 0, stream>>>(erow, ecol, ew, dinv, cur, crow, cnrm, NE);

    // --- GCN stack ---
    static const int cdims[9] = {4, 12, 86, 492, 492, 492, 1896, 1896, 1896};
    static const int cres[8]  = {0, 0, 0, 1, 1, 0, 1, 0};   // residual = layer input
    const float* in = x0;
    float* ping = bufA; float* pong = bufB;
    for (int l = 0; l < 8; ++l) {
        int K = cdims[l], Nc = cdims[l + 1];
        dim3 g((Nc + 127) / 128, (NN + 63) / 64);
        gemm_bf16_wmma<<<g, 256, 0, stream>>>(in, convW[l], nullptr, nullptr, bufH, NN, K, Nc, 0);
        const float* res = cres[l] ? in : nullptr;
        aggregate_kernel<<<NN, 256, 0, stream>>>(bufH, ptr, crow, cnrm, dinv, convB[l], res, ping, Nc);
        in = ping; float* t = ping; ping = pong; pong = t;
    }

    // --- sum-pool per graph ---
    zero_kernel<<<(NG * MAXD + 255) / 256, 256, 0, stream>>>(pool, NG * MAXD);
    pool_kernel<<<NN, 256, 0, stream>>>(in, batch, pool, MAXD);

    // --- FC stack (fused bias + leaky-relu + residual in GEMM epilogue) ---
    static const int fdims[8] = {1896, 1292, 1292, 148, 148, 32, 32, 2};
    static const int fres[7]  = {0, 0, 0, 1, 0, 1, 0};      // residual = layer input
    const float* fin = pool;
    float* fping = fcA; float* fpong = fcBf;
    for (int l = 0; l < 7; ++l) {
        int K = fdims[l], Nc = fdims[l + 1];
        float* outp = (l == 6) ? (float*)d_out : fping;
        dim3 g((Nc + 127) / 128, (NG + 63) / 64);
        const float* res = fres[l] ? fin : nullptr;
        gemm_bf16_wmma<<<g, 256, 0, stream>>>(fin, fcW[l], fcB[l], res, outp, NG, K, Nc, 1);
        fin = outp; float* t = fping; fping = fpong; fpong = t;
    }
}
